// Attention_16784732193285
// MI455X (gfx1250) — matrix operand order repaired and checked by reading, not compiled, not run on hardware
//
#include <hip/hip_runtime.h>
#include <hip/hip_bf16.h>
#include <stdint.h>

typedef __attribute__((ext_vector_type(16))) __bf16 v16bf;
typedef __attribute__((ext_vector_type(8)))  float  v8f;

#define Bsz   32
#define Sdim  2048
#define Hdim  1024
#define TILE_S 64
#define LDA    1032   // padded row stride in halves (2064 B = 129*16, keeps 16B alignment, kills bank conflicts)
#define NB     4      // n-tiles processed concurrently per wave (A-fragment reuse)

__device__ __forceinline__ unsigned short f2bf(float f) {
    union { float f; uint32_t u; } a; a.f = f;
    uint32_t u = a.u;
    uint32_t r = u + 0x7FFFu + ((u >> 16) & 1u);   // RNE
    return (unsigned short)(r >> 16);
}

// Branch-free tanh: prefer hardware V_TANH_F32 (TRANS op on gfx1250).
__device__ __forceinline__ float tanh_fast(float x) {
#if __has_builtin(__builtin_amdgcn_tanhf)
    return __builtin_amdgcn_tanhf(x);
#else
    float ax = __builtin_fabsf(x);
    float e  = __expf(ax + ax);                          // exp(2|x|), inf-safe
    float r  = fmaf(-2.0f, __builtin_amdgcn_rcpf(e + 1.0f), 1.0f);  // rcp(inf)=0 -> r=1
    return __builtin_copysignf(r, x);
#endif
}

union Frag { uint4 q[2]; v16bf v; };

// ---------------- K1: We (= attn_w[:, H:2H]) -> bf16 ----------------
__global__ void convert_we_kernel(const float* __restrict__ attn_w,
                                  unsigned short* __restrict__ WeB) {
    size_t i = (size_t)blockIdx.x * 256 + threadIdx.x;   // over H*H
    int k = (int)(i >> 10);
    int h = (int)(i & 1023);
    WeB[i] = f2bf(attn_w[(size_t)k * (2 * Hdim) + Hdim + h]);
}

// ---------------- K2: pre[b,k] = hidden[b,:] . Wh[k,:] + attn_b[k] ----------------
__global__ void pre_kernel(const float* __restrict__ hidden,
                           const float* __restrict__ attn_w,
                           const float* __restrict__ attn_b,
                           float* __restrict__ pre) {
    int i = blockIdx.x * 256 + threadIdx.x;  // 32*1024
    int b = i >> 10;
    int k = i & 1023;
    const float4* hr = (const float4*)(hidden + (size_t)b * Hdim);
    const float4* wr = (const float4*)(attn_w + (size_t)k * (2 * Hdim)); // Wh row
    float acc = attn_b[k];
    for (int j = 0; j < Hdim / 4; ++j) {
        float4 hv = hr[j], wv = wr[j];
        acc += hv.x * wv.x + hv.y * wv.y + hv.z * wv.z + hv.w * wv.w;
    }
    pre[i] = acc;
}

// ---------------- K3: scores[b,s] = sum_n v[n]*tanh(pre[b,n] + enc[b,s,:].We[n,:]) ----------------
__global__ __launch_bounds__(128)
void attn_scores_kernel(const float* __restrict__ enc,          // [B,S,H] f32
                        const unsigned short* __restrict__ WeB, // [H(n), H(h)] bf16
                        const float* __restrict__ pre,          // [B,H]
                        const float* __restrict__ vw,           // [H]
                        float* __restrict__ scores)             // [B,S]
{
    __shared__ unsigned short sA[TILE_S * LDA];   // 64 x 1024 bf16 enc tile (padded)

    const int b   = blockIdx.y;
    const int s0  = blockIdx.x * TILE_S;
    const int t   = threadIdx.x;
    const int lane = t & 31;
    const int wave = t >> 5;          // 0..3, each owns 16 s-rows

    // ---- stage enc tile -> LDS as bf16 ----
    const float* encb = enc + ((size_t)b * Sdim + s0) * Hdim;
    for (int i = t; i < TILE_S * (Hdim / 4); i += 128) {
        int r  = i / (Hdim / 4);
        int c4 = i % (Hdim / 4);
        float4 f = ((const float4*)(encb + (size_t)r * Hdim))[c4];
        ushort4 u;
        u.x = f2bf(f.x); u.y = f2bf(f.y); u.z = f2bf(f.z); u.w = f2bf(f.w);
        *(ushort4*)(&sA[r * LDA + c4 * 4]) = u;
    }
    __syncthreads();

    const int m0 = wave * 16;         // this wave's s-row base within tile
    const int lg = lane >> 4;         // lane half
    const int lr = lane & 15;         // row (A) / column (B,C) index
    const int kAoff = lg * 8;         // A-fragment k offset per documented layout
    const int kBoff = lg * 8;         // B-fragment k offset per documented layout

    float sacc[8] = {};

    const unsigned short* sArow = &sA[(m0 + lr) * LDA + kAoff];

    for (int nt0 = 0; nt0 < Hdim / 16; nt0 += NB) {
        // 4 concurrent n-tiles: columns n, n+16, n+32, n+48
        const unsigned short* Brow[NB];
        #pragma unroll
        for (int i = 0; i < NB; ++i)
            Brow[i] = WeB + (size_t)((nt0 + i) * 16 + lr) * Hdim + kBoff;

        v8f c[NB];
        #pragma unroll
        for (int i = 0; i < NB; ++i) c[i] = {};

        #pragma unroll 2
        for (int ht = 0; ht < Hdim / 32; ++ht) {
            const int h0 = ht * 32;
            // A 16x32 bf16: lane holds 8 halves at k=kAoff and 8 at k=kAoff+16
            Frag ua;
            ua.q[0] = *(const uint4*)(sArow + h0);
            ua.q[1] = *(const uint4*)(sArow + h0 + 16);
            // B 32x16 bf16: lane = column, holds 8 halves at k=kBoff and 8 at k=kBoff+16
            #pragma unroll
            for (int i = 0; i < NB; ++i) {
                Frag ub;
                ub.q[0] = *(const uint4*)(Brow[i] + h0);
                ub.q[1] = *(const uint4*)(Brow[i] + h0 + 16);
                c[i] = __builtin_amdgcn_wmma_f32_16x16x32_bf16(
                           false, ua.v, false, ub.v, (short)0, c[i], false, false);
            }
        }

        #pragma unroll
        for (int i = 0; i < NB; ++i) {
            const float bias = pre[b * Hdim + (nt0 + i) * 16 + lr];
            const float vn   = vw[(nt0 + i) * 16 + lr];
            #pragma unroll
            for (int j = 0; j < 8; ++j)
                sacc[j] = fmaf(vn, tanh_fast(c[i][j] + bias), sacc[j]);
        }
    }

    // reduce across the 16 lanes that share each C row (they cover the 16 n-columns)
    #pragma unroll
    for (int j = 0; j < 8; ++j) {
        float v = sacc[j];
        #pragma unroll
        for (int off = 1; off < 16; off <<= 1)
            v += __shfl_xor(v, off, 32);
        if (lr == 0)
            scores[(size_t)b * Sdim + s0 + m0 + lg * 8 + j] = v;
    }
}

// ---------------- K4: softmax over S per batch ----------------
__global__ void softmax_kernel(float* __restrict__ scores,     // in: scores, out: normalized (in place)
                               float* __restrict__ attn_out) { // d_out attn region [B,S]
    __shared__ float red[256];
    const int b = blockIdx.x, t = threadIdx.x;
    float* sr = scores + (size_t)b * Sdim;

    float mx = -3.4e38f;
    for (int i = t; i < Sdim; i += 256) mx = fmaxf(mx, sr[i]);
    red[t] = mx; __syncthreads();
    for (int o = 128; o > 0; o >>= 1) {
        if (t < o) red[t] = fmaxf(red[t], red[t + o]);
        __syncthreads();
    }
    const float m = red[0]; __syncthreads();

    float sum = 0.f;
    for (int i = t; i < Sdim; i += 256) sum += __expf(sr[i] - m);
    red[t] = sum; __syncthreads();
    for (int o = 128; o > 0; o >>= 1) {
        if (t < o) red[t] += red[t + o];
        __syncthreads();
    }
    const float inv = 1.f / red[0];

    for (int i = t; i < Sdim; i += 256) {
        float w = __expf(sr[i] - m) * inv;
        attn_out[(size_t)b * Sdim + i] = w;
        sr[i] = w;
    }
}

// ---------------- K5: context[b,h] = sum_s w[b,s]*enc[b,s,h]; concat with hidden ----------------
__global__ void context_kernel(const float* __restrict__ enc,
                               const float* __restrict__ wnorm,
                               const float* __restrict__ hidden,
                               float* __restrict__ out) {       // [B, 2H]
    __shared__ float wsh[256];
    const int b = blockIdx.y;
    const int h = blockIdx.x * 256 + threadIdx.x;
    float acc = 0.f;
    for (int sblk = 0; sblk < Sdim; sblk += 256) {
        wsh[threadIdx.x] = wnorm[(size_t)b * Sdim + sblk + threadIdx.x];
        __syncthreads();
        #pragma unroll 8
        for (int j = 0; j < 256; ++j)
            acc += wsh[j] * enc[((size_t)b * Sdim + sblk + j) * Hdim + h];
        __syncthreads();
    }
    out[(size_t)b * (2 * Hdim) + h]        = acc;
    out[(size_t)b * (2 * Hdim) + Hdim + h] = hidden[(size_t)b * Hdim + h];
}

extern "C" void kernel_launch(void* const* d_in, const int* in_sizes, int n_in,
                              void* d_out, int out_size, void* d_ws, size_t ws_size,
                              hipStream_t stream) {
    const float* hidden = (const float*)d_in[0];
    const float* enc    = (const float*)d_in[1];
    const float* attn_w = (const float*)d_in[2];
    const float* attn_b = (const float*)d_in[3];
    const float* v_w    = (const float*)d_in[4];
    float* out = (float*)d_out;

    // workspace layout
    unsigned short* WeB = (unsigned short*)d_ws;                                     // 2 MB
    float* pre    = (float*)((char*)d_ws + (size_t)2 * 1024 * 1024);                 // 128 KB
    float* scores = (float*)((char*)d_ws + (size_t)2 * 1024 * 1024 + 32 * 1024 * 4); // 256 KB

    convert_we_kernel<<<(Hdim * Hdim) / 256, 256, 0, stream>>>(attn_w, WeB);
    pre_kernel<<<(Bsz * Hdim) / 256, 256, 0, stream>>>(hidden, attn_w, attn_b, pre);

    dim3 g3(Sdim / TILE_S, Bsz);
    attn_scores_kernel<<<g3, 128, 0, stream>>>(enc, WeB, pre, v_w, scores);

    float* attn_out = out + (size_t)Bsz * 2 * Hdim;   // second tuple element
    softmax_kernel<<<Bsz, 256, 0, stream>>>(scores, attn_out);

    dim3 g5(Hdim / 256, Bsz);
    context_kernel<<<g5, 256, 0, stream>>>(enc, scores, hidden, out);
}
